// SCAttention_35373350650693
// MI455X (gfx1250) — matrix operand order repaired and checked by reading, not compiled, not run on hardware
//
#include <hip/hip_runtime.h>

// ============================================================================
// SuperGlue-style attention GNN for MI455X (gfx1250), wave32 + WMMA bf16.
// B=4, C=128, N=2048, K=10 neighbors, H=4 heads (d=32/head).
// Dense math: v_wmma_f32_16x16x32_bf16 (f32 accumulate).
// Weight tiles: global_load_async_to_lds_b128 (ASYNCcnt) straight into LDS.
// Operand fragments: 16B ds_load_b128 from bf16-packed LDS / float4 global.
// ============================================================================

#define NPTS   2048
#define BATCH  4
#define CHNS   128
#define KNN    10
#define NHEADS 4

typedef __bf16 bf16;
typedef bf16  v16bf __attribute__((ext_vector_type(16)));
typedef bf16  v8bf  __attribute__((ext_vector_type(8)));
typedef float v8f   __attribute__((ext_vector_type(8)));

__device__ __forceinline__ v8f v8f_zero() {
  v8f v;
#pragma unroll
  for (int i = 0; i < 8; ++i) v[i] = 0.0f;
  return v;
}

__device__ __forceinline__ v8f wmma_bf16(v16bf a, v16bf b, v8f c) {
  // D = A(16x32 bf16) * B(32x16 bf16) + C(16x16 f32)
  return __builtin_amdgcn_wmma_f32_16x16x32_bf16(false, a, false, b,
                                                 (short)0, c, false, false);
}

__device__ __forceinline__ v16bf cat8(v8bf lo, v8bf hi) {
  return __builtin_shufflevector(lo, hi, 0, 1, 2, 3, 4, 5, 6, 7,
                                 8, 9, 10, 11, 12, 13, 14, 15);
}

// cvt 16 consecutive f32 (4x float4) -> v16bf, with optional scale
__device__ __forceinline__ v16bf cvt16(const float* __restrict__ p, float scale) {
  const float4 c0 = ((const float4*)p)[0];
  const float4 c1 = ((const float4*)p)[1];
  const float4 c2 = ((const float4*)p)[2];
  const float4 c3 = ((const float4*)p)[3];
  const float t[16] = {c0.x, c0.y, c0.z, c0.w, c1.x, c1.y, c1.z, c1.w,
                       c2.x, c2.y, c2.z, c2.w, c3.x, c3.y, c3.z, c3.w};
  v16bf r;
#pragma unroll
  for (int e = 0; e < 16; ++e) r[e] = (bf16)(t[e] * scale);
  return r;
}

// WMMA operand layouts (wave32):
//  A (16x32 bf16): lane row m = lane&15; element e -> k = 16*(e>>3) + 8*(lane>>4) + (e&7)
//  B (32x16 bf16): lane col n = lane&15; element e -> k = 16*(e>>3) + 8*(lane>>4) + (e&7)
//  C/D (16x16 f32): row m = slot + 8*(lane>>4), col n = lane&15.

// ---------------------------------------------------------------------------
// Batched GEMM: Y[b] = Wbf(MxKc,bf16) @ X[b](KcxN,f32) + bias.
// M%64==0, N%64==0, Kc%32==0.  Block=128 (4 waves), tile 64x64, wave 32x32.
//   A tile: async DMA global(bf16) -> LDS  (global_load_async_to_lds_b128)
//   B tile: float4 global -> cvt bf16 -> LDS transposed [n][k]
//   omode 0: Y row-major (m, n).  omode 1: Y head-transposed (h, n, d) with
//            h = m&NHEADS-1, d = m>>2  (for attention Q/K operands).
// ---------------------------------------------------------------------------
__global__ __launch_bounds__(128)
void gemm_bf16_kernel(const bf16* __restrict__ W, const float* __restrict__ X,
                      float* __restrict__ Y, const float* __restrict__ bias,
                      int M, int Kc, int N, long xBS, long yBS, int omode)
{
  // single LDS allocation => base LDS offset 0.
  // [0    .. 4095] : A tile, bf16 [64][32] row-major (m,k)
  // [4096 .. 8191] : B tile, bf16 [64][32] transposed (n,k)
  __shared__ __align__(16) bf16 smem[2][64][32];
  const int b    = blockIdx.z;
  const int m0   = blockIdx.y * 64, n0 = blockIdx.x * 64;
  const int tid  = threadIdx.x;
  const int wave = tid >> 5, lane = tid & 31;
  const int l16  = lane >> 4, lr = lane & 15;
  const int wm   = (wave >> 1) * 32, wn = (wave & 1) * 32;
  const float* Xb = X + (long)b * xBS;
  float*       Yb = Y + (long)b * yBS;
  const unsigned long long wbase = (unsigned long long)W;

  v8f acc[2][2];
#pragma unroll
  for (int i = 0; i < 2; ++i)
#pragma unroll
    for (int j = 0; j < 2; ++j) acc[i][j] = v8f_zero();

  for (int k0 = 0; k0 < Kc; k0 += 32) {
    // --- A tile via async DMA: 64 rows x 64B = 256 x 16B chunks, 2/thread ---
#pragma unroll
    for (int c2 = 0; c2 < 2; ++c2) {
      int chunk = tid * 2 + c2;               // 0..255
      int r = chunk >> 2, c = chunk & 3;      // row, 16B-unit within row
      unsigned ldsoff = (unsigned)(r * 64 + c * 16);                   // bytes
      unsigned goff   = (unsigned)(((m0 + r) * Kc + k0) * 2 + c * 16); // bytes
      asm volatile("global_load_async_to_lds_b128 %0, %1, %2"
                   :: "v"(ldsoff), "v"(goff), "s"(wbase) : "memory");
    }
    // --- B tile: float4 global -> bf16 LDS, transposed to (n,k) ---
#pragma unroll
    for (int it = 0; it < 4; ++it) {
      int i  = tid + it * 128;                // 0..511
      int r  = i >> 4;                        // k row 0..31
      int c4 = (i & 15) * 4;                  // n col group
      float4 xv = *(const float4*)&Xb[(long)(k0 + r) * N + n0 + c4];
      smem[1][c4 + 0][r] = (bf16)xv.x;
      smem[1][c4 + 1][r] = (bf16)xv.y;
      smem[1][c4 + 2][r] = (bf16)xv.z;
      smem[1][c4 + 3][r] = (bf16)xv.w;
    }
    // prefetch next X tile rows into cache while we compute
    if (k0 + 32 < Kc) {
      int r = tid >> 2, c = (tid & 3) * 16;   // 32 rows x 4 chunks
      __builtin_prefetch(&Xb[(long)(k0 + 32 + r) * N + n0 + c], 0, 0);
    }
    asm volatile("s_wait_asynccnt 0x0" ::: "memory");
    __syncthreads();

    v16bf afr[2], bfr[2];
#pragma unroll
    for (int mt = 0; mt < 2; ++mt) {
      int row = wm + 16 * mt + lr;
      v8bf lo = *(const v8bf*)&smem[0][row][8 * l16];
      v8bf hi = *(const v8bf*)&smem[0][row][16 + 8 * l16];
      afr[mt] = cat8(lo, hi);
    }
#pragma unroll
    for (int nt = 0; nt < 2; ++nt) {
      int col = wn + 16 * nt + lr;
      v8bf lo = *(const v8bf*)&smem[1][col][8 * l16];
      v8bf hi = *(const v8bf*)&smem[1][col][16 + 8 * l16];
      bfr[nt] = cat8(lo, hi);
    }
    acc[0][0] = wmma_bf16(afr[0], bfr[0], acc[0][0]);
    acc[0][1] = wmma_bf16(afr[0], bfr[1], acc[0][1]);
    acc[1][0] = wmma_bf16(afr[1], bfr[0], acc[1][0]);
    acc[1][1] = wmma_bf16(afr[1], bfr[1], acc[1][1]);
    __syncthreads();
  }

  // epilogue: bias rows are 8 consecutive floats per mt tile (32B aligned)
#pragma unroll
  for (int mt = 0; mt < 2; ++mt) {
    const int mrow = m0 + wm + 16 * mt + 8 * l16;
    const float4 b0 = *(const float4*)&bias[mrow];
    const float4 b1 = *(const float4*)&bias[mrow + 4];
    const float bv[8] = {b0.x, b0.y, b0.z, b0.w, b1.x, b1.y, b1.z, b1.w};
#pragma unroll
    for (int nt = 0; nt < 2; ++nt) {
      int n = n0 + wn + 16 * nt + lr;
      if (omode == 0) {
        float* yp = Yb + (long)mrow * N + n;
#pragma unroll
        for (int j = 0; j < 8; ++j) yp[(long)j * N] = acc[mt][nt][j] + bv[j];
      } else {
        // (h, n, d) layout: addr = ((m&3)*N + n)*32 + (m>>2)
#pragma unroll
        for (int j = 0; j < 8; ++j) {
          int m = mrow + j;
          Yb[((long)(m & (NHEADS - 1)) * N + n) * 32 + (m >> 2)] =
              acc[mt][nt][j] + bv[j];
        }
      }
    }
  }
}

// ---------------------------------------------------------------------------
// Flash-style multi-head cross attention.
//   QT,KT: (B, H, N, 32) f32 head-transposed (written by gemm omode=1)
//   V,MSG: (B, 128, N) f32 channel-interleaved (channel = d*4+h)
// Wave owns 16 queries, keys in chunks of 32.  Scores: 1 WMMA per 16x16 tile
// (d=32 == WMMA depth).  Online softmax w/ 16-lane shuffle reductions.
// P restriped through bf16 LDS -> two b128 loads per A fragment.
// ---------------------------------------------------------------------------
__global__ __launch_bounds__(128)
void attn_kernel(const float* __restrict__ QT, const float* __restrict__ KT,
                 const float* __restrict__ V, float* __restrict__ MSG, int N)
{
  __shared__ __align__(16) bf16 Pl[4][16][32];
  const int bh = blockIdx.y;
  const int b = bh / NHEADS, h = bh % NHEADS;
  const int wave = threadIdx.x >> 5, lane = threadIdx.x & 31;
  const int l16 = lane >> 4, lr = lane & 15;
  const int n0 = blockIdx.x * 64 + wave * 16;
  const float scale = 0.17677669529663687f; // 1/sqrt(32)
  const long hbase = ((long)b * NHEADS + h) * N * 32;  // QT/KT head base
  const long base  = (long)b * CHNS * N;               // V/MSG batch base

  // Q fragment: lane row n = n0+lr; k(=d) run = [16*l16, +16)
  v16bf aq;
  {
    const float* qrow = QT + hbase + (long)(n0 + lr) * 32;
    v8bf lo, hi;
#pragma unroll
    for (int e = 0; e < 8; ++e) lo[e] = (bf16)(qrow[16 * l16 + e] * scale);
#pragma unroll
    for (int e = 0; e < 8; ++e) hi[e] = (bf16)(qrow[16 * l16 + 8 + e] * scale);
    aq = cat8(lo, hi);
  }

  v8f acc0 = v8f_zero(), acc1 = v8f_zero();
  float mrun[8], lrun[8];
#pragma unroll
  for (int j = 0; j < 8; ++j) { mrun[j] = -1e30f; lrun[j] = 0.0f; }

  for (int mb = 0; mb < N; mb += 32) {
    // K fragments: lane col m; k(=d) run = [16*l16, +16)  -> 64B contiguous
    v16bf bk0 = cvt16(KT + hbase + (long)(mb + lr) * 32 + 16 * l16, 1.0f);
    v16bf bk1 = cvt16(KT + hbase + (long)(mb + 16 + lr) * 32 + 16 * l16, 1.0f);
    v8f s0 = wmma_bf16(aq, bk0, v8f_zero());
    v8f s1 = wmma_bf16(aq, bk1, v8f_zero());

#pragma unroll
    for (int j = 0; j < 8; ++j) {
      float mc = fmaxf(s0[j], s1[j]);
#pragma unroll
      for (int off = 1; off < 16; off <<= 1) mc = fmaxf(mc, __shfl_xor(mc, off, 32));
      float mnew = fmaxf(mrun[j], mc);
      float p0 = __expf(s0[j] - mnew);
      float p1 = __expf(s1[j] - mnew);
      float rs = p0 + p1;
#pragma unroll
      for (int off = 1; off < 16; off <<= 1) rs += __shfl_xor(rs, off, 32);
      float corr = __expf(mrun[j] - mnew);
      lrun[j] = lrun[j] * corr + rs;
      mrun[j] = mnew;
      acc0[j] *= corr;
      acc1[j] *= corr;
      Pl[wave][j + 8 * l16][lr]      = (bf16)p0;
      Pl[wave][j + 8 * l16][16 + lr] = (bf16)p1;
    }

    // P fragment (A layout) from LDS
    v16bf ap;
    {
      v8bf lo = *(const v8bf*)&Pl[wave][lr][16 * l16];
      v8bf hi = *(const v8bf*)&Pl[wave][lr][16 * l16 + 8];
      ap = cat8(lo, hi);
    }
    // V fragments: lane col d = lr (+16); k(=m) run = [16*l16, +16) contiguous
    v16bf bv0 = cvt16(V + base + (long)(lr * NHEADS + h) * N + mb + 16 * l16, 1.0f);
    v16bf bv1 = cvt16(V + base + (long)((lr + 16) * NHEADS + h) * N + mb + 16 * l16, 1.0f);
    acc0 = wmma_bf16(ap, bv0, acc0);
    acc1 = wmma_bf16(ap, bv1, acc1);
  }

#pragma unroll
  for (int j = 0; j < 8; ++j) {
    float inv = 1.0f / lrun[j];
    int n = n0 + j + 8 * l16;
    MSG[base + (long)(lr * NHEADS + h) * N + n]        = acc0[j] * inv;
    MSG[base + (long)((lr + 16) * NHEADS + h) * N + n] = acc1[j] * inv;
  }
}

// ---------------------------------------------------------------------------
// kNN: per point, K nearest (excluding self) by squared distance. coords (B,3,N)
// ---------------------------------------------------------------------------
__global__ __launch_bounds__(128)
void knn_kernel(const float* __restrict__ coords, int* __restrict__ idx, int N)
{
  __shared__ float sx[NPTS], sy[NPTS], sz[NPTS];
  const int b = blockIdx.y;
  const float* cb = coords + (long)b * 3 * N;
  for (int i = threadIdx.x; i < N; i += blockDim.x) {
    sx[i] = cb[i]; sy[i] = cb[N + i]; sz[i] = cb[2 * N + i];
  }
  __syncthreads();
  int n = blockIdx.x * blockDim.x + threadIdx.x;
  if (n >= N) return;
  float px = sx[n], py = sy[n], pz = sz[n];
  float bd[KNN + 1]; int bi[KNN + 1];
#pragma unroll
  for (int j = 0; j <= KNN; ++j) { bd[j] = 1e30f; bi[j] = 0; }
  for (int m = 0; m < N; ++m) {
    float dx = px - sx[m], dy = py - sy[m], dz = pz - sz[m];
    float d = fmaxf(dx * dx + dy * dy + dz * dz, 1e-12f);
    if (d < bd[KNN]) {
      int j = KNN;
      while (j > 0 && bd[j - 1] > d) { bd[j] = bd[j - 1]; bi[j] = bi[j - 1]; --j; }
      bd[j] = d; bi[j] = m;
    }
  }
  for (int j = 0; j < KNN; ++j) idx[((long)b * N + n) * KNN + j] = bi[j + 1];
}

// ---------------------------------------------------------------------------
// Graph-feature helpers.  y[b,o,n,k] = A[b,o,n] + GB[b,o,idx[b,n,k]].
// ---------------------------------------------------------------------------
__global__ __launch_bounds__(256)
void gf_stats_kernel(const float* __restrict__ A, const float* __restrict__ GB,
                     const int* __restrict__ idx, float* __restrict__ mean,
                     float* __restrict__ rstd, int Cout, int N)
{
  __shared__ float rs[256], rs2[256];
  const int o = blockIdx.x, b = blockIdx.y;
  const float* Ab = A  + ((long)b * Cout + o) * N;
  const float* Gb = GB + ((long)b * Cout + o) * N;
  const int*   ib = idx + (long)b * N * KNN;
  const int total = N * KNN;
  float s = 0.0f, s2 = 0.0f;
  for (int t = threadIdx.x; t < total; t += 256) {
    int n = t / KNN, k = t - n * KNN;
    float y = Ab[n] + Gb[ib[(long)n * KNN + k]];
    s += y; s2 += y * y;
  }
  rs[threadIdx.x] = s; rs2[threadIdx.x] = s2;
  __syncthreads();
  for (int w = 128; w > 0; w >>= 1) {
    if ((int)threadIdx.x < w) {
      rs[threadIdx.x]  += rs[threadIdx.x + w];
      rs2[threadIdx.x] += rs2[threadIdx.x + w];
    }
    __syncthreads();
  }
  if (threadIdx.x == 0) {
    float mu = rs[0] / total;
    float var = rs2[0] / total - mu * mu;
    mean[b * Cout + o] = mu;
    rstd[b * Cout + o] = rsqrtf(fmaxf(var, 0.0f) + 1e-5f);
  }
}

__global__ __launch_bounds__(256)
void gf_combine_kernel(const float* __restrict__ A, const float* __restrict__ GB,
                       const int* __restrict__ idx, const float* __restrict__ mean,
                       const float* __restrict__ rstd, float* __restrict__ out,
                       long outBS, int Cout, int N)
{
  const int n = blockIdx.x * 256 + threadIdx.x;
  const int o = blockIdx.y, b = blockIdx.z;
  const float* Ab = A  + ((long)b * Cout + o) * N;
  const float* Gb = GB + ((long)b * Cout + o) * N;
  const int*   ib = idx + ((long)b * N + n) * KNN;
  float mx = -1e30f;
#pragma unroll
  for (int k = 0; k < KNN; ++k) mx = fmaxf(mx, Gb[ib[k]]);
  float z = (Ab[n] + mx - mean[b * Cout + o]) * rstd[b * Cout + o];
  out[(long)b * outBS + (long)o * N + n] = z > 0.0f ? z : 0.2f * z;
}

// ---------------------------------------------------------------------------
// Instance norm over N (+ ReLU or leaky-ReLU 0.2), in place.  grid=(C,B).
// ---------------------------------------------------------------------------
__global__ __launch_bounds__(256)
void inorm_act_kernel(float* __restrict__ Y, long bs, int N, int relu)
{
  __shared__ float rs[256], rs2[256];
  const int c = blockIdx.x, b = blockIdx.y;
  float* p = Y + (long)b * bs + (long)c * N;
  float s = 0.0f, s2 = 0.0f;
  for (int i = threadIdx.x; i < N; i += 256) { float v = p[i]; s += v; s2 += v * v; }
  rs[threadIdx.x] = s; rs2[threadIdx.x] = s2;
  __syncthreads();
  for (int w = 128; w > 0; w >>= 1) {
    if ((int)threadIdx.x < w) {
      rs[threadIdx.x]  += rs[threadIdx.x + w];
      rs2[threadIdx.x] += rs2[threadIdx.x + w];
    }
    __syncthreads();
  }
  float mu = rs[0] / N;
  float var = rs2[0] / N - mu * mu;
  float r = rsqrtf(fmaxf(var, 0.0f) + 1e-5f);
  for (int i = threadIdx.x; i < N; i += 256) {
    float z = (p[i] - mu) * r;
    p[i] = relu ? fmaxf(z, 0.0f) : (z > 0.0f ? z : 0.2f * z);
  }
}

// ---------------------------------------------------------------------------
// Small elementwise helpers
// ---------------------------------------------------------------------------
__global__ void split_w_kernel(const float* __restrict__ W, bf16* __restrict__ Wd,
                               bf16* __restrict__ Wn, int Cout, int Hh)
{
  int i = blockIdx.x * 256 + threadIdx.x;
  if (i < Cout * Hh) {
    int o = i / Hh, c = i - o * Hh;
    float wc = W[(long)o * 2 * Hh + c];
    float wn = W[(long)o * 2 * Hh + Hh + c];
    Wd[i] = (bf16)(wc - wn);   // center contribution (Wc - Wn)
    Wn[i] = (bf16)wn;          // neighbor weight
  }
}

__global__ void cvt_bf16_kernel(bf16* __restrict__ dst, const float* __restrict__ src, long n)
{
  long i = (long)blockIdx.x * 256 + threadIdx.x;
  if (i < n) dst[i] = (bf16)src[i];
}

__global__ void fillz_kernel(float* __restrict__ dst, long n)
{
  long i = (long)blockIdx.x * 256 + threadIdx.x;
  if (i < n) dst[i] = 0.0f;
}

__global__ void copyf_kernel(float* __restrict__ dst, const float* __restrict__ src, long n)
{
  long i = (long)blockIdx.x * 256 + threadIdx.x;
  if (i < n) dst[i] = src[i];
}

__global__ void copy_ch_kernel(const float* __restrict__ src, long sBS,
                               float* __restrict__ dst, long dBS, long cnt)
{
  long i = (long)blockIdx.x * 256 + threadIdx.x;
  int b = blockIdx.z;
  if (i < cnt) dst[(long)b * dBS + i] = src[(long)b * sBS + i];
}

__global__ void add_kernel(float* __restrict__ d, const float* __restrict__ s, long n)
{
  long i = (long)blockIdx.x * 256 + threadIdx.x;
  if (i < n) d[i] += s[i];
}

// ===========================================================================
// Host orchestration
// ===========================================================================
extern "C" void kernel_launch(void* const* d_in, const int* in_sizes, int n_in,
                              void* d_out, int out_size, void* d_ws, size_t ws_size,
                              hipStream_t stream)
{
  (void)in_sizes; (void)n_in; (void)out_size; (void)ws_size;
  const int B = BATCH, N = NPTS, H = NHEADS;
  const long NL = N;
  const size_t NF = (size_t)B * CHNS * N; // 1,048,576 floats

  const float* desc0   = (const float*)d_in[0];
  const float* desc1   = (const float*)d_in[1];
  const float* coords0 = (const float*)d_in[2];
  const float* coords1 = (const float*)d_in[3];

  // params flattened as jax pytree leaves (dicts sorted by key):
  // cross0(4..15): kb,kw,m1b,m1w,m2b,m2w,mb,mw,qb,qw,vb,vw
  // cross1(16..27), self0(28..30): w1,w2,w3, self1(31..33)
  struct CrossP { const float *kb,*kw,*m1b,*m1w,*m2b,*m2w,*mb,*mw,*qb,*qw,*vb,*vw; };
  struct SelfP  { const float *w1,*w2,*w3; };
  auto getCross = [&](int base) {
    CrossP p;
    p.kb  = (const float*)d_in[base + 0];  p.kw  = (const float*)d_in[base + 1];
    p.m1b = (const float*)d_in[base + 2];  p.m1w = (const float*)d_in[base + 3];
    p.m2b = (const float*)d_in[base + 4];  p.m2w = (const float*)d_in[base + 5];
    p.mb  = (const float*)d_in[base + 6];  p.mw  = (const float*)d_in[base + 7];
    p.qb  = (const float*)d_in[base + 8];  p.qw  = (const float*)d_in[base + 9];
    p.vb  = (const float*)d_in[base + 10]; p.vw  = (const float*)d_in[base + 11];
    return p;
  };
  auto getSelf = [&](int base) {
    SelfP p;
    p.w1 = (const float*)d_in[base + 0];
    p.w2 = (const float*)d_in[base + 1];
    p.w3 = (const float*)d_in[base + 2];
    return p;
  };
  CrossP cross0 = getCross(4), cross1 = getCross(16);
  SelfP  self0  = getSelf(28), self1  = getSelf(31);

  // workspace carve (256B-aligned regions)
  char* wp = (char*)d_ws;
  auto carve = [&](size_t bytes) {
    void* r = (void*)wp;
    wp += (bytes + 255) & ~(size_t)255;
    return r;
  };
  float* D0   = (float*)carve(NF * 4);          // B*128*N
  float* D1   = (float*)carve(NF * 4);
  float* X3   = (float*)carve(4 * NF * 4);      // B*512*N (Q/K/V/MSG in cross)
  float* TA   = (float*)carve(2 * NF * 4);      // B*256*N
  float* TB   = (float*)carve(2 * NF * 4);
  float* MEAN = (float*)carve(2048 * 4);
  float* RSTD = (float*)carve(2048 * 4);
  float* ZB   = (float*)carve(256 * 4);         // zero bias
  bf16*  WDb  = (bf16*)carve(32768 * 2);        // split weights (bf16)
  bf16*  WNb  = (bf16*)carve(32768 * 2);
  bf16*  WBF  = (bf16*)carve(65536 * 2);        // generic bf16 weight scratch
  int*   I0   = (int*)carve((size_t)B * N * KNN * 4);
  int*   I1   = (int*)carve((size_t)B * N * KNN * 4);

  auto gemm = [&](const bf16* Wp, const float* Xp, float* Yp, const float* bias,
                  int M, int Kc, long xBS, long yBS, int omode = 0) {
    dim3 g(N / 64, M / 64, B), blk(128);
    gemm_bf16_kernel<<<g, blk, 0, stream>>>(Wp, Xp, Yp, bias ? bias : ZB,
                                            M, Kc, N, xBS, yBS, omode);
  };
  auto cvtw = [&](const float* src, long n) {
    cvt_bf16_kernel<<<dim3((unsigned)((n + 255) / 256)), 256, 0, stream>>>(WBF, src, n);
    return (const bf16*)WBF;
  };

  auto self_attn = [&](float* D, const int* idx, SelfP sp) {
    // stage 1: graph conv w1 (128 <- 2*128) -> x1 in X3[:,128:256,:]
    split_w_kernel<<<dim3((128 * 128 + 255) / 256), 256, 0, stream>>>(sp.w1, WDb, WNb, 128, 128);
    gemm(WDb, D, TA, nullptr, 128, 128, 128 * NL, 128 * NL); // A  = (Wc-Wn)@x
    gemm(WNb, D, TB, nullptr, 128, 128, 128 * NL, 128 * NL); // GB = Wn@x
    gf_stats_kernel<<<dim3(128, B), 256, 0, stream>>>(TA, TB, idx, MEAN, RSTD, 128, N);
    gf_combine_kernel<<<dim3(N / 256, 128, B), 256, 0, stream>>>(
        TA, TB, idx, MEAN, RSTD, X3 + (size_t)128 * N, 512 * NL, 128, N);
    // x0 -> X3[:,0:128,:]
    copy_ch_kernel<<<dim3((unsigned)((128 * NL) / 256), 1, B), 256, 0, stream>>>(
        D, 128 * NL, X3, 512 * NL, 128 * NL);
    // stage 2: graph conv w2 (256 <- 2*128) -> x2 in X3[:,256:512,:]
    split_w_kernel<<<dim3((256 * 128 + 255) / 256), 256, 0, stream>>>(sp.w2, WDb, WNb, 256, 128);
    gemm(WDb, X3 + (size_t)128 * N, TA, nullptr, 256, 128, 512 * NL, 256 * NL);
    gemm(WNb, X3 + (size_t)128 * N, TB, nullptr, 256, 128, 512 * NL, 256 * NL);
    gf_stats_kernel<<<dim3(256, B), 256, 0, stream>>>(TA, TB, idx, MEAN, RSTD, 256, N);
    gf_combine_kernel<<<dim3(N / 256, 256, B), 256, 0, stream>>>(
        TA, TB, idx, MEAN, RSTD, X3 + (size_t)256 * N, 512 * NL, 256, N);
    // stage 3: w3 (128 <- 512) over concat, then inorm + leaky relu -> D
    gemm(cvtw(sp.w3, 128 * 512), X3, D, nullptr, 128, 512, 512 * NL, 128 * NL);
    inorm_act_kernel<<<dim3(128, B), 256, 0, stream>>>(D, 128 * NL, N, 0);
  };

  auto cross_attn = [&](float* Dt, const float* Ds, CrossP cp) {
    float* QT  = X3;            // (B,H,N,32) head-transposed
    float* KT  = X3 + NF;       // (B,H,N,32) head-transposed
    float* V   = X3 + 2 * NF;   // (B,128,N)  channel-interleaved
    float* MSG = X3 + 3 * NF;
    gemm(cvtw(cp.qw, 16384), Dt, QT, cp.qb, 128, 128, 128 * NL, 128 * NL, 1);
    gemm(cvtw(cp.kw, 16384), Ds, KT, cp.kb, 128, 128, 128 * NL, 128 * NL, 1);
    gemm(cvtw(cp.vw, 16384), Ds, V,  cp.vb, 128, 128, 128 * NL, 128 * NL, 0);
    attn_kernel<<<dim3(N / 64, B * H), 128, 0, stream>>>(QT, KT, V, MSG, N);
    // h = concat([x, mproj]) in TA (B,256,N)
    copy_ch_kernel<<<dim3((unsigned)((128 * NL) / 256), 1, B), 256, 0, stream>>>(
        Dt, 128 * NL, TA, 256 * NL, 128 * NL);
    gemm(cvtw(cp.mw, 16384), MSG, TA + (size_t)128 * N, cp.mb, 128, 128, 128 * NL, 256 * NL);
    gemm(cvtw(cp.m1w, 65536), TA, TB, cp.m1b, 256, 256, 256 * NL, 256 * NL);
    inorm_act_kernel<<<dim3(256, B), 256, 0, stream>>>(TB, 256 * NL, N, 1);
    gemm(cvtw(cp.m2w, 32768), TB, MSG, cp.m2b, 128, 256, 256 * NL, 128 * NL); // delta
    add_kernel<<<dim3((unsigned)(NF / 256)), 256, 0, stream>>>(Dt, MSG, (long)NF);
  };

  // init: zero bias, working descriptors, kNN (coords constant across layers)
  fillz_kernel<<<dim3(1), 256, 0, stream>>>(ZB, 256);
  copyf_kernel<<<dim3((unsigned)(NF / 256)), 256, 0, stream>>>(D0, desc0, (long)NF);
  copyf_kernel<<<dim3((unsigned)(NF / 256)), 256, 0, stream>>>(D1, desc1, (long)NF);
  knn_kernel<<<dim3(N / 128, B), 128, 0, stream>>>(coords0, I0, N);
  knn_kernel<<<dim3(N / 128, B), 128, 0, stream>>>(coords1, I1, N);

  // layer pipeline (matches reference sequencing, incl. cross using updated D0)
  self_attn(D0, I0, self0);
  self_attn(D1, I1, self0);
  cross_attn(D0, D1, cross0);
  cross_attn(D1, D0, cross0);
  self_attn(D0, I0, self1);
  self_attn(D1, I1, self1);
  cross_attn(D0, D1, cross1);
  cross_attn(D1, D0, cross1);

  // outputs concatenated: (desc0, desc1)
  float* out = (float*)d_out;
  copyf_kernel<<<dim3((unsigned)(NF / 256)), 256, 0, stream>>>(out, D0, (long)NF);
  copyf_kernel<<<dim3((unsigned)(NF / 256)), 256, 0, stream>>>(out + NF, D1, (long)NF);
}
